// SpectralNet_75909251989848
// MI455X (gfx1250) — matrix operand order repaired and checked by reading, not compiled, not run on hardware
//
#include <hip/hip_runtime.h>
#include <math.h>

// ---------------------------------------------------------------------------
// Types for CDNA5 WMMA (wave32): v_wmma_f32_16x16x32_bf16
// ---------------------------------------------------------------------------
typedef __attribute__((ext_vector_type(16))) __bf16 v16bf;
typedef __attribute__((ext_vector_type(8)))  float  v8f;

union alignas(32) FragBf {
    uint4 q[2];
    v16bf v;
};

static constexpr int NPTS   = 65536;
static constexpr int KNBR   = 16;
static constexpr int IN_DIM = 128;
static constexpr int HID    = 1024;
static constexpr int HID2   = 512;
static constexpr int OUTD   = 32;
static constexpr int BKT    = 64;   // K elements staged per LDS tile (2 WMMA k-steps)

// ---------------------------------------------------------------------------
// CDNA5 async global->LDS copy (ASYNCcnt), 16 bytes per lane per issue.
// vdst = LDS byte address (low 32 bits of flat LDS pointer), vaddr = 64-bit
// global address. No VGPR data round-trip, no per-chunk waits.
// ---------------------------------------------------------------------------
__device__ inline void async_load_b128(unsigned lds_off, const void* gptr) {
    asm volatile("global_load_async_to_lds_b128 %0, %1, off"
                 :: "v"(lds_off), "v"(gptr) : "memory");
}

__device__ inline void wait_async0() {
#if __has_builtin(__builtin_amdgcn_s_wait_asynccnt)
    __builtin_amdgcn_s_wait_asynccnt(0);
#else
    asm volatile("s_wait_asynccnt 0x0" ::: "memory");
#endif
}

// ---------------------------------------------------------------------------
// LDS fragment loads in contiguous k order (k = 16*half + s). LDS tiles are
// [rows][BKT] bf16 row-major (A: rows = M, B: rows = N since W is
// pre-transposed to [N][K]); ks selects the 32-wide K sub-step.
// A frag: lane l<16 -> row base+l, K ks+0..15;
//         lane l>=16 -> same rows, K ks+16..31.
// B frag: lane l<16 -> col base+l, K ks+0..15; lane l>=16 -> K ks+16..31.
// ---------------------------------------------------------------------------
__device__ inline v16bf load_a_frag(const __bf16* As, int rowBase, int ks, int lane) {
    int half = lane >> 4;
    const __bf16* p = As + (rowBase + (lane & 15)) * BKT + ks;
    FragBf f;
    f.q[0] = *(const uint4*)(p + half * 16);
    f.q[1] = *(const uint4*)(p + half * 16 + 8);
    return f.v;
}

__device__ inline v16bf load_b_frag(const __bf16* Bs, int colBase, int ks, int lane) {
    int half = lane >> 4;
    const __bf16* p = Bs + (colBase + (lane & 15)) * BKT + ks + half * 16;
    FragBf f;
    f.q[0] = *(const uint4*)(p);
    f.q[1] = *(const uint4*)(p + 8);
    return f.v;
}

// ---------------------------------------------------------------------------
// Double-buffered async-pipelined bf16 GEMM:
//   C[M,Ntot] = act(A[M,K] * Bt[Ntot,K]^T + bias)
// Block tile BM x BN, K-stage 64, MW*NW waves, wave computes WMT x WNT 16x16.
// ---------------------------------------------------------------------------
template <int BM, int BN, int MW, int NW, int WMT, int WNT, bool RELU, bool OUTBF>
__global__ __launch_bounds__(MW * NW * 32)
void gemm_bf16_kernel(const __bf16* __restrict__ A,
                      const __bf16* __restrict__ Bt,
                      const float*  __restrict__ bias,
                      void* __restrict__ Cout,
                      int M, int Ntot, int K)
{
    constexpr int NT  = MW * NW * 32;
    constexpr int ACH = (BM * BKT / 8) / NT;   // 16B chunks per thread (A tile)
    constexpr int BCH = (BN * BKT / 8) / NT;   // 16B chunks per thread (B tile)
    static_assert(ACH >= 1 && BCH >= 1, "tile/thread mismatch");

    __shared__ __bf16 As[2][BM * BKT];
    __shared__ __bf16 Bs[2][BN * BKT];

    const int tid  = threadIdx.x;
    const int lane = tid & 31;
    const int wave = tid >> 5;
    const int wm   = wave % MW;
    const int wn   = wave / MW;
    const int m0   = blockIdx.y * BM;
    const int n0   = blockIdx.x * BN;

    auto issue_stage = [&](int buf, int k0) {
        #pragma unroll
        for (int it = 0; it < ACH; ++it) {
            int c = tid + it * NT;                 // 8 chunks per 64-elem row
            int r = c >> 3, kq = (c & 7) * 8;
            async_load_b128((unsigned)(uintptr_t)&As[buf][r * BKT + kq],
                            &A[(size_t)(m0 + r) * K + k0 + kq]);
        }
        #pragma unroll
        for (int it = 0; it < BCH; ++it) {
            int c = tid + it * NT;
            int r = c >> 3, kq = (c & 7) * 8;
            async_load_b128((unsigned)(uintptr_t)&Bs[buf][r * BKT + kq],
                            &Bt[(size_t)(n0 + r) * K + k0 + kq]);
        }
    };

    v8f acc[WMT][WNT] = {};

    issue_stage(0, 0);
    const int ktiles = K / BKT;
    for (int kt = 0; kt < ktiles; ++kt) {
        const int buf = kt & 1;
        wait_async0();          // my async writes into `buf` are done
        __syncthreads();        // everyone's are; tile `buf` is ready
        if (kt + 1 < ktiles) issue_stage(buf ^ 1, (kt + 1) * BKT);

        #pragma unroll
        for (int ks = 0; ks < BKT; ks += 32) {
            v16bf af[WMT], bfr[WNT];
            #pragma unroll
            for (int t = 0; t < WMT; ++t)
                af[t] = load_a_frag(&As[buf][0], (wm * WMT + t) * 16, ks, lane);
            #pragma unroll
            for (int u = 0; u < WNT; ++u)
                bfr[u] = load_b_frag(&Bs[buf][0], (wn * WNT + u) * 16, ks, lane);
            #pragma unroll
            for (int t = 0; t < WMT; ++t)
                #pragma unroll
                for (int u = 0; u < WNT; ++u)
                    acc[t][u] = __builtin_amdgcn_wmma_f32_16x16x32_bf16(
                        false, af[t], false, bfr[u], (short)0, acc[t][u],
                        false, false);
        }
        // Reads of `buf` are consumed by the WMMAs above before this wave can
        // signal the next top-of-loop barrier, and the next writes into `buf`
        // are only issued after that barrier -> one barrier per stage is safe.
    }

    // Epilogue: C layout (16x16 f32): VGPR r, lanes 0-15 -> M=r, lanes 16-31 -> M=r+8
    #pragma unroll
    for (int t = 0; t < WMT; ++t) {
        #pragma unroll
        for (int u = 0; u < WNT; ++u) {
            int mbase = m0 + (wm * WMT + t) * 16 + (lane >> 4) * 8;
            int nidx  = n0 + (wn * WNT + u) * 16 + (lane & 15);
            float bv  = bias[nidx];
            #pragma unroll
            for (int r = 0; r < 8; ++r) {
                float v = acc[t][u][r] + bv;
                if (RELU) v = fmaxf(v, 0.0f);
                size_t off = (size_t)(mbase + r) * Ntot + nidx;
                if (OUTBF) ((__bf16*)Cout)[off] = (__bf16)v;
                else       ((float*)Cout)[off]  = v;
            }
        }
    }
}

// ---------------------------------------------------------------------------
// Small helper kernels
// ---------------------------------------------------------------------------
__global__ void zero_f32_kernel(float* p, int n) {
    int i = blockIdx.x * blockDim.x + threadIdx.x;
    if (i < n) p[i] = 0.0f;
}

__global__ void cvt_f32_bf16_kernel(const float* __restrict__ x,
                                    __bf16* __restrict__ y, int n) {
    int i = blockIdx.x * blockDim.x + threadIdx.x;
    if (i < n) y[i] = (__bf16)x[i];
}

// Wt[n][k] = bf16(W[k][n])  (W stored [K][N] row-major)
__global__ void transpose_cvt_kernel(const float* __restrict__ W,
                                     __bf16* __restrict__ Wt, int K, int N) {
    int idx = blockIdx.x * blockDim.x + threadIdx.x;
    if (idx < K * N) {
        int k = idx / N, n = idx % N;
        Wt[(size_t)n * K + k] = (__bf16)W[idx];
    }
}

// d[i] += 0.5*sum_k w  and scatter 0.5*w to d[nn]; w = exp(-d^2/(2 sigma^2))
__global__ void degree_kernel(const float* __restrict__ dists,
                              const int* __restrict__ nn,
                              float* __restrict__ d, int n) {
    int i = blockIdx.x * blockDim.x + threadIdx.x;
    if (i >= n) return;
    const float* dr = dists + (size_t)i * KNBR;
    float s = 0.0f;
    #pragma unroll
    for (int k = 0; k < KNBR; ++k) s += dr[k];
    float sigma  = s * (1.0f / KNBR);
    float inv2s2 = 1.0f / (2.0f * sigma * sigma);
    float local = 0.0f;
    #pragma unroll
    for (int k = 0; k < KNBR; ++k) {
        float dd = dr[k];
        float wh = 0.5f * __expf(-dd * dd * inv2s2);
        local += wh;
        atomicAdd(&d[nn[(size_t)i * KNBR + k]], wh);
    }
    atomicAdd(&d[i], local);
}

// gram[32][32] += (Y/d)^T (Y/d), one wave per row-chunk, shuffle rank-1 update
__global__ void gram_kernel(const float* __restrict__ Y,
                            const float* __restrict__ d,
                            float* __restrict__ gram, int n) {
    int lane = threadIdx.x & 31;
    int wave = (blockIdx.x * blockDim.x + threadIdx.x) >> 5;
    int nw   = (gridDim.x * blockDim.x) >> 5;
    float acc[OUTD];
    #pragma unroll
    for (int i = 0; i < OUTD; ++i) acc[i] = 0.0f;
    for (int r = wave; r < n; r += nw) {
        float v = Y[(size_t)r * OUTD + lane] / d[r];
        #pragma unroll
        for (int i = 0; i < OUTD; ++i)
            acc[i] += __shfl(v, i, 32) * v;
    }
    #pragma unroll
    for (int i = 0; i < OUTD; ++i)
        atomicAdd(&gram[i * OUTD + lane], acc[i]);
}

// 32x32 Cholesky of (gram + 1e-7 I), operator = inv(L)^T * sqrtN. One wave.
__global__ void chol_op_kernel(const float* __restrict__ gram,
                               float* __restrict__ op, float sqrtN) {
    __shared__ float G[OUTD][OUTD];
    __shared__ float L[OUTD][OUTD];
    int t = threadIdx.x;
    for (int i = 0; i < OUTD; ++i)
        G[i][t] = gram[i * OUTD + t] + ((i == t) ? 1e-7f : 0.0f);
    __syncthreads();
    for (int k = 0; k < OUTD; ++k) {
        if (t == k) L[k][k] = sqrtf(G[k][k]);
        __syncthreads();
        if (t > k)       L[t][k] = G[t][k] / L[k][k];
        else if (t < k)  L[t][k] = 0.0f;
        __syncthreads();
        if (t > k)
            for (int j = k + 1; j <= t; ++j)
                G[t][j] -= L[t][k] * L[j][k];
        __syncthreads();
    }
    // forward substitution: thread t solves column t of inv(L)
    float x[OUTD];
    for (int i = 0; i < OUTD; ++i) {
        float s = (i == t) ? 1.0f : 0.0f;
        for (int j = t; j < i; ++j) s -= L[i][j] * x[j];
        x[i] = (i >= t) ? s / L[i][i] : 0.0f;
    }
    // op[a][b] = inv(L)[b][a] * sqrtN  -> thread t writes row t
    for (int i = 0; i < OUTD; ++i) op[t * OUTD + i] = x[i] * sqrtN;
}

// Z[r] = (Y[r] @ op) / d2[r], one wave per row (shuffle broadcast of Y row)
__global__ void apply_op_kernel(const float* __restrict__ Y,
                                const float* __restrict__ op,
                                const float* __restrict__ d,
                                float* __restrict__ Z, int n) {
    __shared__ float ops[OUTD * OUTD];
    int tid = threadIdx.x;
    for (int i = tid; i < OUTD * OUTD; i += blockDim.x) ops[i] = op[i];
    __syncthreads();
    int lane = tid & 31;
    int wave = (blockIdx.x * blockDim.x + tid) >> 5;
    int nw   = (gridDim.x * blockDim.x) >> 5;
    for (int r = wave; r < n; r += nw) {
        float y = Y[(size_t)r * OUTD + lane];
        float s = 0.0f;
        #pragma unroll
        for (int k = 0; k < OUTD; ++k)
            s += __shfl(y, k, 32) * ops[k * OUTD + lane];
        Z[(size_t)r * OUTD + lane] = s / d[r];
    }
}

// loss = (1/N) * sum_edges ||z_i - z_nn||^2 * (d2[i] + d2[nn]); warp per edge
__global__ void loss_kernel(const float* __restrict__ Z,
                            const int* __restrict__ nn,
                            const float* __restrict__ d2,
                            float* __restrict__ loss, int n, float invN) {
    int lane   = threadIdx.x & 31;
    int wave   = (blockIdx.x * blockDim.x + threadIdx.x) >> 5;
    int nwaves = (gridDim.x * blockDim.x) >> 5;
    int nedges = n * KNBR;
    float acc = 0.0f;
    for (int e = wave; e < nedges; e += nwaves) {
        int i = e >> 4;        // K = 16
        int j = nn[e];
        float diff = Z[(size_t)i * OUTD + lane] - Z[(size_t)j * OUTD + lane];
        float de = diff * diff;
        #pragma unroll
        for (int off = 16; off; off >>= 1) de += __shfl_xor(de, off, 32);
        if (lane == 0) acc += de * (d2[i] + d2[j]);
    }
    if (lane == 0) atomicAdd(loss, acc * invN);
}

// ---------------------------------------------------------------------------
// Host orchestration
// ---------------------------------------------------------------------------
static void run_mapper(const __bf16* Xb, const __bf16* W1t, const __bf16* W2t,
                       const __bf16* W3t, const __bf16* W4t,
                       const float* b1, const float* b2, const float* b3, const float* b4,
                       __bf16* H1, __bf16* H2, float* Y, hipStream_t stream)
{
    // L1: [N,128] x [128,1024] -> H1 (relu, bf16)
    hipLaunchKernelGGL((gemm_bf16_kernel<128,128,4,2,2,4,true,true>),
                       dim3(HID/128, NPTS/128), dim3(256), 0, stream,
                       Xb, W1t, b1, (void*)H1, NPTS, HID, IN_DIM);
    // L2: [N,1024] x [1024,1024] -> H2 (relu, bf16)
    hipLaunchKernelGGL((gemm_bf16_kernel<128,128,4,2,2,4,true,true>),
                       dim3(HID/128, NPTS/128), dim3(256), 0, stream,
                       H1, W2t, b2, (void*)H2, NPTS, HID, HID);
    // L3: [N,1024] x [1024,512] -> H3 (reuse H1 region) (relu, bf16)
    hipLaunchKernelGGL((gemm_bf16_kernel<128,128,4,2,2,4,true,true>),
                       dim3(HID2/128, NPTS/128), dim3(256), 0, stream,
                       H2, W3t, b3, (void*)H1, NPTS, HID2, HID);
    // L4: [N,512] x [512,32] -> Y (no relu, f32)
    hipLaunchKernelGGL((gemm_bf16_kernel<128,32,8,1,1,2,false,false>),
                       dim3(OUTD/32, NPTS/128), dim3(256), 0, stream,
                       H1, W4t, b4, (void*)Y, NPTS, OUTD, HID2);
}

extern "C" void kernel_launch(void* const* d_in, const int* in_sizes, int n_in,
                              void* d_out, int out_size, void* d_ws, size_t ws_size,
                              hipStream_t stream)
{
    const float* x1     = (const float*)d_in[0];
    const float* x2     = (const float*)d_in[1];
    const float* dists1 = (const float*)d_in[2];
    const int*   nnidx1 = (const int*)  d_in[3];
    const float* dists2 = (const float*)d_in[4];
    const int*   nnidx2 = (const int*)  d_in[5];
    const float* W1 = (const float*)d_in[6];  const float* b1 = (const float*)d_in[7];
    const float* W2 = (const float*)d_in[8];  const float* b2 = (const float*)d_in[9];
    const float* W3 = (const float*)d_in[10]; const float* b3 = (const float*)d_in[11];
    const float* W4 = (const float*)d_in[12]; const float* b4 = (const float*)d_in[13];

    // ---- workspace carve-up (all 256B aligned) ----
    char* ws = (char*)d_ws;
    size_t off = 0;
    auto alloc = [&](size_t bytes) { char* p = ws + off; off += (bytes + 255) & ~(size_t)255; return p; };

    __bf16* Xb  = (__bf16*)alloc((size_t)NPTS * IN_DIM * 2);   // 16 MiB
    __bf16* W1t = (__bf16*)alloc((size_t)HID * IN_DIM * 2);
    __bf16* W2t = (__bf16*)alloc((size_t)HID * HID * 2);
    __bf16* W3t = (__bf16*)alloc((size_t)HID2 * HID * 2);
    __bf16* W4t = (__bf16*)alloc((size_t)OUTD * HID2 * 2);
    __bf16* H1  = (__bf16*)alloc((size_t)NPTS * HID * 2);      // 128 MiB (H3 reuses)
    __bf16* H2  = (__bf16*)alloc((size_t)NPTS * HID * 2);      // 128 MiB
    float*  Y   = (float*) alloc((size_t)NPTS * OUTD * 4);     // 8 MiB
    float*  Z   = (float*) alloc((size_t)NPTS * OUTD * 4);     // 8 MiB
    float*  d1  = (float*) alloc((size_t)NPTS * 4);
    float*  d2  = (float*) alloc((size_t)NPTS * 4);
    float*  gram= (float*) alloc(OUTD * OUTD * 4);
    float*  op  = (float*) alloc(OUTD * OUTD * 4);
    (void)ws_size; (void)in_sizes; (void)n_in; (void)out_size;

    // ---- weight transpose + bf16 conversion ----
    auto launch_tr = [&](const float* W, __bf16* Wt, int K, int N) {
        int tot = K * N;
        hipLaunchKernelGGL(transpose_cvt_kernel, dim3((tot + 255) / 256), dim3(256),
                           0, stream, W, Wt, K, N);
    };
    launch_tr(W1, W1t, IN_DIM, HID);
    launch_tr(W2, W2t, HID,    HID);
    launch_tr(W3, W3t, HID,    HID2);
    launch_tr(W4, W4t, HID2,   OUTD);

    // ---- degrees ----
    hipLaunchKernelGGL(zero_f32_kernel, dim3((NPTS + 255) / 256), dim3(256), 0, stream, d1, NPTS);
    hipLaunchKernelGGL(zero_f32_kernel, dim3((NPTS + 255) / 256), dim3(256), 0, stream, d2, NPTS);
    hipLaunchKernelGGL(degree_kernel, dim3((NPTS + 255) / 256), dim3(256), 0, stream,
                       dists1, nnidx1, d1, NPTS);
    hipLaunchKernelGGL(degree_kernel, dim3((NPTS + 255) / 256), dim3(256), 0, stream,
                       dists2, nnidx2, d2, NPTS);

    // ---- pass 1: mapper(x1), gram, cholesky/operator ----
    hipLaunchKernelGGL(cvt_f32_bf16_kernel, dim3((NPTS * IN_DIM + 255) / 256), dim3(256),
                       0, stream, x1, Xb, NPTS * IN_DIM);
    run_mapper(Xb, W1t, W2t, W3t, W4t, b1, b2, b3, b4, H1, H2, Y, stream);

    hipLaunchKernelGGL(zero_f32_kernel, dim3((OUTD * OUTD + 255) / 256), dim3(256),
                       0, stream, gram, OUTD * OUTD);
    hipLaunchKernelGGL(gram_kernel, dim3(512), dim3(256), 0, stream, Y, d1, gram, NPTS);
    hipLaunchKernelGGL(chol_op_kernel, dim3(1), dim3(32), 0, stream, gram, op, 256.0f /* sqrt(65536) */);

    // ---- pass 2: mapper(x2), apply operator, edge loss ----
    hipLaunchKernelGGL(cvt_f32_bf16_kernel, dim3((NPTS * IN_DIM + 255) / 256), dim3(256),
                       0, stream, x2, Xb, NPTS * IN_DIM);
    run_mapper(Xb, W1t, W2t, W3t, W4t, b1, b2, b3, b4, H1, H2, Y, stream);

    hipLaunchKernelGGL(apply_op_kernel, dim3(512), dim3(256), 0, stream, Y, op, d2, Z, NPTS);

    hipLaunchKernelGGL(zero_f32_kernel, dim3(1), dim3(32), 0, stream, (float*)d_out, 1);
    hipLaunchKernelGGL(loss_kernel, dim3(1024), dim3(256), 0, stream,
                       Z, nnidx2, d2, (float*)d_out, NPTS, 1.0f / (float)NPTS);
}
